// SerializedAttention_26654567039603
// MI455X (gfx1250) — compile-verified, matrix operand order checked by reading, not run on hardware
//
#include <hip/hip_runtime.h>
#include <hip/hip_bf16.h>

// ---------------------------------------------------------------------------
// MI455X (gfx1250) fused serialized-attention.
//   K0: f32 -> bf16 conversion (x, qkv_w, proj_w)
//   K1: per-patch fused  QKV-GEMM -> attention (softmax(QK^T/8)V)  [WMMA bf16]
//       Q/K/V for all 8 heads live in the 320KB WGP LDS; x tile staged with
//       GLOBAL_LOAD_ASYNC_TO_LDS_B128 (ASYNCcnt-tracked) when available.
//   K2: output projection GEMM (bf16 WMMA, f32 out)
// ---------------------------------------------------------------------------

typedef __attribute__((ext_vector_type(16))) __bf16 v16bf;
typedef __attribute__((ext_vector_type(8)))  __bf16 v8bf;
typedef __attribute__((ext_vector_type(8)))  float  v8f;
typedef __attribute__((ext_vector_type(4)))  int    vi4;

#define NPTS 131072
#define CCH  512
#define NHEAD 8
#define PATCH 64
#define HDIM 64
#define XLD  520   // padded LDS row for x tile (16B aligned, avoids conflicts)

#if defined(__has_builtin)
#if __has_builtin(__builtin_amdgcn_global_load_async_to_lds_b128)
#define HAVE_ASYNC_LDS 1
#endif
#endif

#define AS_GLOBAL_I4(p) ((__attribute__((address_space(1))) vi4*)(p))
#define AS_LDS_I4(p)    ((__attribute__((address_space(3))) vi4*)(p))

__device__ __forceinline__ v8f wmma_bf16(v16bf a, v16bf b, v8f c) {
  // D = A(16x32,bf16) x B(32x16,bf16) + C(16x16,f32)
  return __builtin_amdgcn_wmma_f32_16x16x32_bf16(
      /*neg_a=*/false, a, /*neg_b=*/false, b,
      /*c_mod=*/(short)0, c, /*reuse_a=*/false, /*reuse_b=*/false);
}

// A/B fragment loader (ISA 7.12.2 16-bit layouts): lane (l&15) picks the
// row (A) / column-as-weight-row (B); each lane loads two contiguous 16B
// chunks: K = [k0+8h .. +7] and [k0+16+8h .. +7], h = lane>>4.
__device__ __forceinline__ v16bf load_frag_k(const __bf16* base, int ld,
                                             int r0, int k0) {
  const int lane = threadIdx.x & 31;
  const __bf16* p = base + (size_t)(r0 + (lane & 15)) * ld
                         + (k0 + ((lane >> 4) << 3));
  union { v16bf v; v8bf h[2]; } u;
  u.h[0] = *(const v8bf*)(p);
  u.h[1] = *(const v8bf*)(p + 16);
  return u.v;
}

// C/D tile (16x16 f32) -> bf16 store: VGPR v holds row r0+v+8*(lane>>4),
// col = c0 + (lane&15).
__device__ __forceinline__ void store_tile_bf16(__bf16* dst, int ld,
                                                int r0, int c0, v8f a) {
  const int lane = threadIdx.x & 31;
  const int col = c0 + (lane & 15);
  const int rb  = r0 + ((lane >> 4) << 3);
#pragma unroll
  for (int v = 0; v < 8; ++v)
    dst[(size_t)(rb + v) * ld + col] = (__bf16)a[v];
}

// Transposed store (lays V down as V^T[d][q] so P@V B-frags are contiguous
// LDS rows).
__device__ __forceinline__ void store_tile_bf16_T(__bf16* dst, int ld,
                                                  int r0, int c0, v8f a) {
  const int lane = threadIdx.x & 31;
  const int col = c0 + (lane & 15);
  const int rb  = r0 + ((lane >> 4) << 3);
#pragma unroll
  for (int v = 0; v < 8; ++v)
    dst[(size_t)col * ld + (rb + v)] = (__bf16)a[v];
}

// ------------------------------- K0 ----------------------------------------
__global__ void cvt_f32_bf16(const float* __restrict__ src,
                             __bf16* __restrict__ dst, long n) {
  long i = (long)blockIdx.x * blockDim.x + threadIdx.x;
  long stride = (long)gridDim.x * blockDim.x;
  for (; i < n; i += stride) dst[i] = (__bf16)src[i];
}

// ------------------------------- K1 ----------------------------------------
// One workgroup (256 thr = 8 waves) per 64-point patch.
// LDS: Qs[8][64][64] | Ks[8][64][64] | Vt[8][64][64] | Xt[64][520] (~257KB)
// Pb (per-head 64x64 probs) reuses the Xt region after the QKV phase.
__global__ __launch_bounds__(256) void fused_qkv_attn(
    const __bf16* __restrict__ xb,     // (N, 512) bf16
    const __bf16* __restrict__ wqkv,   // (1536, 512) bf16
    const float*  __restrict__ bqkv,   // (1536) f32
    __bf16* __restrict__ ao)           // (N, 512) bf16 attention output
{
  extern __shared__ __bf16 smem[];
  __bf16* Qs = smem;                       // 8*64*64
  __bf16* Ks = Qs + NHEAD * PATCH * HDIM;
  __bf16* Vt = Ks + NHEAD * PATCH * HDIM;  // stored transposed: [h][d][q]
  __bf16* Xt = Vt + NHEAD * PATCH * HDIM;  // 64 x XLD
  __bf16* Pb = Xt;                         // reuse for probabilities

  const int tid  = threadIdx.x;
  const int wave = tid >> 5;
  const int lane = tid & 31;
  const size_t row0 = (size_t)blockIdx.x * PATCH;

  // ---- stage x tile (64 x 512 bf16) into LDS: 4 threads per row ----
  {
    const int r = tid >> 2;
    const int c = (tid & 3) * 128;
    const __bf16* src = xb + (row0 + r) * CCH + c;
    __bf16* dst = Xt + r * XLD + c;
#if HAVE_ASYNC_LDS
#pragma unroll
    for (int i = 0; i < 128; i += 8)
      __builtin_amdgcn_global_load_async_to_lds_b128(
          AS_GLOBAL_I4(src + i), AS_LDS_I4(dst + i), /*offset=*/0, /*cpol=*/0);
#if __has_builtin(__builtin_amdgcn_s_wait_asynccnt)
    __builtin_amdgcn_s_wait_asynccnt(0);
#else
    asm volatile("s_wait_asynccnt 0" ::: "memory");
#endif
#else
#pragma unroll
    for (int i = 0; i < 128; i += 8)
      *(v8bf*)(dst + i) = *(const v8bf*)(src + i);
#endif
  }
  __syncthreads();

  // ---- QKV GEMM: Y(64 x 1536) = Xt @ wqkv^T + b; wave owns 192 columns.
  //      Two 16-col tiles per pass: 8 WMMAs per k-step for 4 A-frag loads.
#pragma unroll 1
  for (int np = 0; np < 6; ++np) {
    const int c0a = wave * 192 + np * 32;   // tiles never straddle s/h bounds
    v8f acc[2][4] = {};
#pragma unroll 1
    for (int kt = 0; kt < 16; ++kt) {
      const int k0 = kt * 32;
      v16bf b0 = load_frag_k(wqkv, CCH, c0a,      k0);  // weight rows (L2 hot)
      v16bf b1 = load_frag_k(wqkv, CCH, c0a + 16, k0);
#pragma unroll
      for (int mt = 0; mt < 4; ++mt) {
        v16bf afr = load_frag_k(Xt, XLD, mt * 16, k0);
        acc[0][mt] = wmma_bf16(afr, b0, acc[0][mt]);
        acc[1][mt] = wmma_bf16(afr, b1, acc[1][mt]);
      }
    }
#pragma unroll
    for (int j = 0; j < 2; ++j) {
      const int c0g = c0a + j * 16;
      const int s   = c0g >> 9;        // 0=q 1=k 2=v
      const int rem = c0g & 511;
      const int h   = rem >> 6;
      const int d0  = rem & 63;
      const float bias = bqkv[c0g + (lane & 15)];
#pragma unroll
      for (int mt = 0; mt < 4; ++mt) {
        v8f a = acc[j][mt];
#pragma unroll
        for (int v = 0; v < 8; ++v) a[v] += bias;
        if (s == 0)      store_tile_bf16  (Qs + h * 4096, HDIM, mt * 16, d0, a);
        else if (s == 1) store_tile_bf16  (Ks + h * 4096, HDIM, mt * 16, d0, a);
        else             store_tile_bf16_T(Vt + h * 4096, PATCH, mt * 16, d0, a);
      }
    }
  }
  __syncthreads();

  // ---- attention: wave == head ----
  const __bf16* Qh = Qs + wave * 4096;
  const __bf16* Kh = Ks + wave * 4096;
  const __bf16* Vh = Vt + wave * 4096;    // [d][q]
  __bf16* Ph = Pb + wave * 4096;          // [p][q] probs, wave-private

  // S = Q K^T  (64x64, K-dim 64)
  v8f S[4][4] = {};
#pragma unroll
  for (int kt = 0; kt < 2; ++kt) {
    const int k0 = kt * 32;
    v16bf bfr[4];
#pragma unroll
    for (int nt = 0; nt < 4; ++nt) bfr[nt] = load_frag_k(Kh, HDIM, nt * 16, k0);
#pragma unroll
    for (int mt = 0; mt < 4; ++mt) {
      v16bf afr = load_frag_k(Qh, HDIM, mt * 16, k0);
#pragma unroll
      for (int nt = 0; nt < 4; ++nt)
        S[mt][nt] = wmma_bf16(afr, bfr[nt], S[mt][nt]);
    }
  }

  // row softmax in registers: row = (mt*16 + v + 8*half); its 64 cols are
  // the 4 nt tiles x 16 lanes of this half. xor 1/2/4/8 stays in-half.
  const float scale = 0.125f;  // 64^-0.5
#pragma unroll
  for (int mt = 0; mt < 4; ++mt) {
#pragma unroll
    for (int v = 0; v < 8; ++v) {
      float t0 = S[mt][0][v] * scale, t1 = S[mt][1][v] * scale;
      float t2 = S[mt][2][v] * scale, t3 = S[mt][3][v] * scale;
      float m = fmaxf(fmaxf(t0, t1), fmaxf(t2, t3));
#pragma unroll
      for (int off = 8; off >= 1; off >>= 1)
        m = fmaxf(m, __shfl_xor(m, off, 32));
      float e0 = __expf(t0 - m), e1 = __expf(t1 - m);
      float e2 = __expf(t2 - m), e3 = __expf(t3 - m);
      float sum = e0 + e1 + e2 + e3;
#pragma unroll
      for (int off = 8; off >= 1; off >>= 1)
        sum += __shfl_xor(sum, off, 32);
      float inv = 1.0f / sum;
      S[mt][0][v] = e0 * inv; S[mt][1][v] = e1 * inv;
      S[mt][2][v] = e2 * inv; S[mt][3][v] = e3 * inv;
    }
  }

  // round-trip P through LDS to get A-fragment layout
#pragma unroll
  for (int mt = 0; mt < 4; ++mt)
#pragma unroll
    for (int nt = 0; nt < 4; ++nt)
      store_tile_bf16(Ph, PATCH, mt * 16, nt * 16, S[mt][nt]);

  // O = P @ V   (B-frags are contiguous rows of V^T)
  v8f O[4][4] = {};
#pragma unroll
  for (int kt = 0; kt < 2; ++kt) {
    const int k0 = kt * 32;
    v16bf bfr[4];
#pragma unroll
    for (int nt = 0; nt < 4; ++nt) bfr[nt] = load_frag_k(Vh, PATCH, nt * 16, k0);
#pragma unroll
    for (int mt = 0; mt < 4; ++mt) {
      v16bf afr = load_frag_k(Ph, PATCH, mt * 16, k0);
#pragma unroll
      for (int nt = 0; nt < 4; ++nt)
        O[mt][nt] = wmma_bf16(afr, bfr[nt], O[mt][nt]);
    }
  }

  __bf16* dsto = ao + row0 * CCH + wave * HDIM;
#pragma unroll
  for (int mt = 0; mt < 4; ++mt)
#pragma unroll
    for (int nt = 0; nt < 4; ++nt)
      store_tile_bf16(dsto, CCH, mt * 16, nt * 16, O[mt][nt]);
}

// ------------------------------- K2 ----------------------------------------
// out(N,512) = ao(bf16) @ proj_w^T + proj_b, f32 output.
// Block = 64x64 tile, 8 waves x 2 tiles, K-loop 512/32.
__global__ __launch_bounds__(256) void proj_gemm(
    const __bf16* __restrict__ ao, const __bf16* __restrict__ wp,
    const float* __restrict__ bp, float* __restrict__ out)
{
  const int mb = blockIdx.x * 64;
  const int nb = blockIdx.y * 64;
  const int wave = threadIdx.x >> 5;
  const int lane = threadIdx.x & 31;
  const int t0 = wave * 2;
  const int mt = t0 >> 2;             // tiles t0, t0+1 share the row tile
  const int nt0 = t0 & 3, nt1 = (t0 + 1) & 3;

  v8f acc0 = {}, acc1 = {};
#pragma unroll 1
  for (int kt = 0; kt < 16; ++kt) {
    const int k0 = kt * 32;
    // prefetch next K-slice of the HBM-streamed activation operand
    if (kt + 1 < 16)
      __builtin_prefetch(ao + (size_t)(mb + mt * 16 + (lane & 15)) * CCH
                            + k0 + 32, 0, 0);
    v16bf afr = load_frag_k(ao, CCH, mb + mt * 16, k0);
    v16bf b0  = load_frag_k(wp, CCH, nb + nt0 * 16, k0);
    v16bf b1  = load_frag_k(wp, CCH, nb + nt1 * 16, k0);
    acc0 = wmma_bf16(afr, b0, acc0);
    acc1 = wmma_bf16(afr, b1, acc1);
  }
  const int rb = mb + mt * 16 + ((lane >> 4) << 3);
  const int c0 = nb + nt0 * 16 + (lane & 15);
  const int c1 = nb + nt1 * 16 + (lane & 15);
  const float bi0 = bp[c0], bi1 = bp[c1];
#pragma unroll
  for (int v = 0; v < 8; ++v) {
    out[(size_t)(rb + v) * CCH + c0] = acc0[v] + bi0;
    out[(size_t)(rb + v) * CCH + c1] = acc1[v] + bi1;
  }
}

// ------------------------------ launch -------------------------------------
extern "C" void kernel_launch(void* const* d_in, const int* in_sizes, int n_in,
                              void* d_out, int out_size, void* d_ws, size_t ws_size,
                              hipStream_t stream) {
  (void)in_sizes; (void)n_in; (void)out_size; (void)ws_size;
  const float* x      = (const float*)d_in[0];
  // d_in[1]: batch indices (int64) — unused: scenes are patch-aligned
  const float* qkv_w  = (const float*)d_in[2];
  const float* qkv_b  = (const float*)d_in[3];
  const float* proj_w = (const float*)d_in[4];
  const float* proj_b = (const float*)d_in[5];
  float* out = (float*)d_out;

  char* ws = (char*)d_ws;
  __bf16* xb   = (__bf16*)ws; ws += (size_t)NPTS * CCH * 2;       // 134 MB
  __bf16* ao   = (__bf16*)ws; ws += (size_t)NPTS * CCH * 2;       // 134 MB
  __bf16* wq   = (__bf16*)ws; ws += (size_t)3 * CCH * CCH * 2;    // 1.5 MB
  __bf16* wp   = (__bf16*)ws; ws += (size_t)CCH * CCH * 2;        // 0.5 MB

  cvt_f32_bf16<<<2048, 256, 0, stream>>>(x, xb, (long)NPTS * CCH);
  cvt_f32_bf16<<<96, 256, 0, stream>>>(qkv_w, wq, (long)3 * CCH * CCH);
  cvt_f32_bf16<<<32, 256, 0, stream>>>(proj_w, wp, (long)CCH * CCH);

  const size_t smem = (size_t)(3 * NHEAD * PATCH * HDIM + PATCH * XLD)
                      * sizeof(__bf16);   // ~257 KB of the 320 KB WGP LDS
  fused_qkv_attn<<<NPTS / PATCH, 256, smem, stream>>>(xb, wq, qkv_b, ao);

  dim3 g2(NPTS / 64, CCH / 64);
  proj_gemm<<<g2, 256, 0, stream>>>(ao, wp, proj_b, out);
}
